// GeodesicLoss_27032524161557
// MI455X (gfx1250) — matrix operand order repaired and checked by reading, not compiled, not run on hardware
//
#include <hip/hip_runtime.h>
#include <hip/hip_bf16.h>
#include <math.h>

typedef __attribute__((ext_vector_type(16))) _Float16 v16h;
typedef __attribute__((ext_vector_type(8)))  _Float16 v8h;
typedef __attribute__((ext_vector_type(8)))  float    v8f;

#define GEO_STEPS 10
#define GEO_D 32

// LDS layout:
//   G region: Gamma as f16, [j][i][k] with 80-byte row pitch (32 f16 = 64B + 16B pad)
//   T region: per-wave 16x32 f16 scratch for vel transpose (C-layout -> A-layout)
static constexpr int GPITCH = 80;                 // bytes per (j,i) row, 16B aligned
static constexpr int G_BYTES = 32 * 32 * GPITCH;  // 81920
static constexpr int T_BYTES_PER_WAVE = 16 * 32 * 2; // 1024
static constexpr int SMEM_BYTES = G_BYTES + 8 * T_BYTES_PER_WAVE; // 90112 (<320KB/WGP)

__global__ void geo_zero_kernel(float* out) {
    if (threadIdx.x == 0 && blockIdx.x == 0) out[0] = 0.0f;
}

__global__ __launch_bounds__(256)
void GeodesicLoss_27032524161557_kernel(const float* __restrict__ outputs,
                                        const float* __restrict__ targets,
                                        const float* __restrict__ gamma,
                                        float* __restrict__ out, int Btot) {
    extern __shared__ char smem[];
    const int tid  = threadIdx.x;
    const int lane = tid & 31;
    const int wave = tid >> 5;
    const int half = lane >> 4;    // 0: lanes 0-15, 1: lanes 16-31
    const int l16  = lane & 15;
    const int m    = half ? 8 : 0; // C/D-layout row offset for this half

    // ---- Stage Gamma[i][j][k] (f32, global) -> G_lds[j][i][k] (f16, 80B pitch) ----
    // Gamma is shared by every block: leave it with default (RT) cache policy so it
    // stays L2-resident; the batch streams below use NT hints instead.
    for (int idx = tid; idx < 32 * 32 * 32; idx += blockDim.x) {
        int k = idx & 31;
        int j = (idx >> 5) & 31;
        int i = idx >> 10;
        _Float16 v = (_Float16)gamma[idx];   // idx == (i*32 + j)*32 + k
        *(_Float16*)(smem + (j * 32 + i) * GPITCH + k * 2) = v;
    }
    __syncthreads();

    const int base = blockIdx.x * 128 + wave * 16; // batch tile base for this wave
    char* T = smem + G_BYTES + wave * T_BYTES_PER_WAVE;

    // ---- State in WMMA C/D layout: two N-tiles (dims 0-15 and 16-31), f32 ----
    v8f traj0, traj1, vel0, vel1;
#pragma unroll
    for (int r = 0; r < 8; ++r) {
        int b = base + m + r;
        traj0[r] = __builtin_nontemporal_load(&outputs[b * 32 + l16]);
        traj1[r] = __builtin_nontemporal_load(&outputs[b * 32 + 16 + l16]);
        vel0[r]  = 0.0f;
        vel1[r]  = 0.0f;
    }

    const float dt = 1.0f / GEO_STEPS;
#pragma unroll 1
    for (int s = 0; s < GEO_STEPS; ++s) {
        // vel (C-layout regs) -> T[b16][d32] row-major f16
#pragma unroll
        for (int r = 0; r < 8; ++r) {
            int b = m + r;
            *(_Float16*)(T + (b * 32 + l16) * 2)        = (_Float16)vel0[r];
            *(_Float16*)(T + (b * 32 + 16 + l16) * 2)   = (_Float16)vel1[r];
        }
        __syncthreads();

        // A-layout vel row fragment: lane row M = l16
        // half 0: k in [0,16) ; half 1: k in [16,32)
        const char* Trow = T + l16 * 64; // 32 f16 = 64B per row
        v8h c0 = *(const v8h*)(Trow + 32 * half);
        v8h c1 = *(const v8h*)(Trow + 16 + 32 * half);
        v16h Af;
#pragma unroll
        for (int e = 0; e < 8; ++e) { Af[e] = c0[e]; Af[8 + e] = c1[e]; }

        v8f acc0 = {}; v8f acc1 = {};
        // K = j*32 + k ; K-tile kt == j. A-frag(kt) = vel[b][kt] * Af.
        // Vector multiply -> v_pk_mul_f16 (8 packed muls, co-executable with WMMA).
#pragma unroll
        for (int kt4 = 0; kt4 < 32; kt4 += 4) {
            v16h a[4];
#pragma unroll
            for (int u = 0; u < 4; ++u) {
                _Float16 sj = *(const _Float16*)(Trow + (kt4 + u) * 2);
                a[u] = Af * sj;   // packed f16: 8x v_pk_mul_f16
            }
#pragma unroll
            for (int u = 0; u < 4; ++u) {
                // B frags: B[k][i] = Gamma[i][kt][k]; lane col i = l16 (+16 for ntile1),
                // k range 16*half..16*half+15 -> 32B = 2x ds_load_b128
                const char* g0 = smem + ((kt4 + u) * 32 + l16) * GPITCH + 32 * half;
                const char* g1 = g0 + 16 * GPITCH;
                v16h b0 = *(const v16h*)g0;
                v16h b1 = *(const v16h*)g1;
                acc0 = __builtin_amdgcn_wmma_f32_16x16x32_f16(false, a[u], false, b0,
                                                              (short)0, acc0, false, false);
                acc1 = __builtin_amdgcn_wmma_f32_16x16x32_f16(false, a[u], false, b1,
                                                              (short)0, acc1, false, false);
            }
        }
        __syncthreads(); // T rewritten next step

        // acc_i = -(w @ G) ; vel += acc*dt ; traj += vel*dt  (all layout-aligned)
#pragma unroll
        for (int r = 0; r < 8; ++r) {
            vel0[r] -= acc0[r] * dt;
            vel1[r] -= acc1[r] * dt;
            traj0[r] += vel0[r] * dt;
            traj1[r] += vel1[r] * dt;
        }
    }

    // ---- mean_b || traj_b - target_b ||_2 : reduce dims across 16-lane halves ----
    float distsum = 0.0f;
#pragma unroll
    for (int r = 0; r < 8; ++r) {
        int b = base + m + r;
        float t0 = __builtin_nontemporal_load(&targets[b * 32 + l16]);
        float t1 = __builtin_nontemporal_load(&targets[b * 32 + 16 + l16]);
        float d0 = traj0[r] - t0;
        float d1 = traj1[r] - t1;
        float e = d0 * d0 + d1 * d1;
        e += __shfl_xor(e, 1, 32);
        e += __shfl_xor(e, 2, 32);
        e += __shfl_xor(e, 4, 32);
        e += __shfl_xor(e, 8, 32);   // all 16 lanes of the half now hold sum for row b
        distsum += sqrtf(e);
    }
    distsum += __shfl_xor(distsum, 16, 32); // combine lower/upper half row groups
    if (lane == 0) atomicAdd(out, distsum * (1.0f / (float)Btot));
}

extern "C" void kernel_launch(void* const* d_in, const int* in_sizes, int n_in,
                              void* d_out, int out_size, void* d_ws, size_t ws_size,
                              hipStream_t stream) {
    const float* outputs = (const float*)d_in[0];
    const float* targets = (const float*)d_in[1];
    const float* gamma   = (const float*)d_in[2];
    float* out = (float*)d_out;

    const int Btot = in_sizes[0] / GEO_D; // 524288
    geo_zero_kernel<<<1, 64, 0, stream>>>(out);

    const int blocks = Btot / 128; // 128 batch rows per 256-thread block (8 waves x 16)
    GeodesicLoss_27032524161557_kernel<<<blocks, 256, SMEM_BYTES, stream>>>(
        outputs, targets, gamma, out, Btot);
}
